// MaskedSelfAttention_24386824307382
// MI455X (gfx1250) — matrix operand order repaired and checked by reading, not compiled, not run on hardware
//
#include <hip/hip_runtime.h>
#include <hip/hip_bf16.h>
#include <math.h>

// ---------------------------------------------------------------------------
// MaskedSelfAttention on gfx1250 (MI455X), bf16 WMMA path + TDM staging.
// B=8, L=2048, DIM_X=1024, DIM_ATT=2048 (hardcoded per reference).
// ---------------------------------------------------------------------------

typedef __attribute__((ext_vector_type(16))) __bf16       v16bf;
typedef __attribute__((ext_vector_type(8)))  float        v8f;
typedef __attribute__((ext_vector_type(4)))  unsigned int uint4v;
typedef __attribute__((ext_vector_type(2)))  unsigned int uint2v;
typedef __attribute__((ext_vector_type(4)))  float        float4v;
typedef __attribute__((ext_vector_type(4)))  unsigned int u32x4;
typedef __attribute__((ext_vector_type(8)))  int          i32x8;
typedef __attribute__((ext_vector_type(4)))  int          i32x4;

#define BATCH   8
#define SEQ     2048
#define DIMX    1024
#define DIMA    2048
#define LDSK    40          // padded LDS row stride in halfwords (80 B, 16B-aligned)

union BF16Frag { uint4v q[2]; v16bf v; };   // 32 bytes = 8 VGPRs

__device__ __forceinline__ unsigned short f32_to_bf16(float f) {
  union { float f; unsigned int u; } c; c.f = f;
  unsigned int u = c.u;
  u = u + 0x7FFFu + ((u >> 16) & 1u);       // round-to-nearest-even
  return (unsigned short)(u >> 16);
}

__device__ __forceinline__ unsigned int pack_bf16x2(float a, float b) {
  return (unsigned int)f32_to_bf16(a) | ((unsigned int)f32_to_bf16(b) << 16);
}

// A fragment: 16x32 bf16 subtile, row-major (stride LDSK) in LDS.
// Lane holds M=lane&15; K runs {g*8..+7} and {16+g*8..+7}, g=lane>>4 (ISA 7.12.2).
__device__ __forceinline__ v16bf load_frag_A(const unsigned short* sub, int lane) {
  const int m = lane & 15, g = lane >> 4;
  const unsigned short* row = sub + m * LDSK;
  BF16Frag f;
  f.q[0] = *(const uint4v*)(row + g * 8);
  f.q[1] = *(const uint4v*)(row + 16 + g * 8);
  return f.v;
}

// B fragment from a TRANSPOSED tile (N-major: [n][k], stride LDSK).
// Lane holds N=lane&15; K runs {g*8..+7} and {16+g*8..+7}, g=lane>>4.
__device__ __forceinline__ v16bf load_frag_B(const unsigned short* sub, int lane) {
  const int n = lane & 15, g = lane >> 4;
  const unsigned short* row = sub + n * LDSK;
  BF16Frag f;
  f.q[0] = *(const uint4v*)(row + g * 8);
  f.q[1] = *(const uint4v*)(row + 16 + g * 8);
  return f.v;
}

// ---------------------------------------------------------------------------
// TDM: load a 128-row x 64-byte tile (128 x 32 bf16, viewed as 16 u32/row)
// from a row-major bf16 matrix with row stride DIMA into LDS with an 80-byte
// padded row stride (pad_interval: 16 dwords -> code 3; pad_amount: 4 dwords
// -> code 3).  D# layout per CDNA5 ISA.
// This toolchain exposes the 6-arg builtin:
//   (uint32x4 g0, int32x8 g1, int32x4 g2, int32x4 g3, int32x8 pad, i32 cpol)
// ---------------------------------------------------------------------------
__device__ __forceinline__ void tdm_load_tile(const unsigned short* gtile,
                                              unsigned int lds_byte_off) {
  const unsigned long long ga = (unsigned long long)(uintptr_t)gtile;
  u32x4 g0 = {};
  g0.x = 1u;                                             // count=1 (valid user D#)
  g0.y = lds_byte_off;                                   // lds_addr [63:32]
  g0.z = (unsigned int)ga;                               // global_addr[31:0]
  g0.w = (unsigned int)((ga >> 32) & 0x01FFFFFFull)      // global_addr[56:32]
         | 0x80000000u;                                  // type=2 ("image")
  i32x8 g1 = {};
  g1[0] = (2 << 16)        // data_size = 4 bytes
        | (1 << 20)        // pad_enable
        | (3 << 22)        // pad_interval: 16 dwords
        | (3 << 25);       // pad_amount:   4 dwords
  g1[1] = (int)((DIMA / 2) << 16);        // tensor_dim0[15:0] (u32 units), bits 79:48
  g1[2] = (int)(16384u << 16);            // tensor_dim1[15:0], bits 111:80
  g1[3] = (int)(16u << 16);               // tile_dim0 = 16 u32 (64 B), bits 127:112
  g1[4] = 128;                            // tile_dim1 = 128 rows (bits 143:128)
  g1[5] = DIMA / 2;                       // tensor_dim0_stride[31:0] = 1024 u32
  g1[6] = 0;
  g1[7] = 0;
  i32x4 z4 = {};
  i32x8 z8 = {};
  __builtin_amdgcn_tensor_load_to_lds(g0, g1, z4, z4, z8, 0);
}

// ---------------------------------------------------------------------------
// Kernel 1: fused QKV projection GEMM. (B*L,1024) x (1024,2048) per weight.
// blockIdx.z selects q/k/v. q,k stored bf16 for the scores GEMM; v stored f32.
// Macro tile 128M x 128N, K-step 32, 8 waves = 2(M) x 4(N).
// ---------------------------------------------------------------------------
__global__ __launch_bounds__(256)
void qkv_proj_kernel(const float* __restrict__ x,
                     const float* __restrict__ wq,
                     const float* __restrict__ wk,
                     const float* __restrict__ wv,
                     unsigned short* __restrict__ q_bf,
                     unsigned short* __restrict__ k_bf,
                     float* __restrict__ v_f32) {
  const int n0 = blockIdx.x * 128;
  const int m0 = blockIdx.y * 128;
  const int which = blockIdx.z;
  const float* w = (which == 0) ? wq : (which == 1) ? wk : wv;

  __shared__ __align__(16) unsigned short ldsA[128 * LDSK];   // x tile [m][k]
  __shared__ __align__(16) unsigned short ldsB[128 * LDSK];   // w tile transposed [n][k]

  const int tid  = threadIdx.x;
  const int lane = tid & 31;
  const int wave = tid >> 5;
  const int wm   = wave & 1;    // 64-row half
  const int wn   = wave >> 1;   // 32-col slice

  v8f zero = {};
  v8f acc[4][2];
#pragma unroll
  for (int mi = 0; mi < 4; ++mi)
#pragma unroll
    for (int ni = 0; ni < 2; ++ni) acc[mi][ni] = zero;

  for (int k0 = 0; k0 < DIMX; k0 += 32) {
    __syncthreads();
    // Stage x: 128x32 f32 -> bf16, packed pairs, 8-byte LDS stores.
#pragma unroll
    for (int p = 0; p < 4; ++p) {
      const int e   = tid + p * 256;      // 1024 chunks of 4 floats
      const int row = e >> 3;
      const int ch  = e & 7;
      const float4v xv = *(const float4v*)(x + (size_t)(m0 + row) * DIMX + k0 + ch * 4);
      uint2v pk;
      pk.x = pack_bf16x2(xv.x, xv.y);
      pk.y = pack_bf16x2(xv.z, xv.w);
      *(uint2v*)(ldsA + row * LDSK + ch * 4) = pk;
    }
    // Stage w transposed: two consecutive k per thread -> one b32 LDS store.
#pragma unroll
    for (int p = 0; p < 8; ++p) {
      const int e  = tid + p * 256;       // 2048 packed pairs
      const int n  = e & 127;
      const int kk = e >> 7;              // 0..15 -> k = 2*kk
      const float wa = w[(size_t)(k0 + 2 * kk)     * DIMA + n0 + n];
      const float wb = w[(size_t)(k0 + 2 * kk + 1) * DIMA + n0 + n];
      *(unsigned int*)(ldsB + n * LDSK + 2 * kk) = pack_bf16x2(wa, wb);
    }
    __syncthreads();

    v16bf afr[4], bfr[2];
#pragma unroll
    for (int mi = 0; mi < 4; ++mi)
      afr[mi] = load_frag_A(ldsA + (wm * 64 + mi * 16) * LDSK, lane);
#pragma unroll
    for (int ni = 0; ni < 2; ++ni)
      bfr[ni] = load_frag_B(ldsB + (wn * 32 + ni * 16) * LDSK, lane);
#pragma unroll
    for (int mi = 0; mi < 4; ++mi)
#pragma unroll
      for (int ni = 0; ni < 2; ++ni)
        acc[mi][ni] = __builtin_amdgcn_wmma_f32_16x16x32_bf16(
            false, afr[mi], false, bfr[ni], (short)0, acc[mi][ni], false, false);
  }

  // Epilogue: D layout = lane gives N=lane&15; VGPR r gives M = r + (lane>>4)*8.
  // `which` is block-uniform: hoist the branch out of the store loops.
  const int nb = n0 + wn * 32 + (lane & 15);
  const int mb = m0 + wm * 64 + ((lane >> 4) << 3);
  if (which == 2) {
#pragma unroll
    for (int mi = 0; mi < 4; ++mi)
#pragma unroll
      for (int ni = 0; ni < 2; ++ni)
#pragma unroll
        for (int r = 0; r < 8; ++r)
          v_f32[(size_t)(mb + mi * 16 + r) * DIMA + (nb + ni * 16)] = acc[mi][ni][r];
  } else {
    unsigned short* dst = (which == 0) ? q_bf : k_bf;
#pragma unroll
    for (int mi = 0; mi < 4; ++mi)
#pragma unroll
      for (int ni = 0; ni < 2; ++ni)
#pragma unroll
        for (int r = 0; r < 8; ++r)
          dst[(size_t)(mb + mi * 16 + r) * DIMA + (nb + ni * 16)] =
              f32_to_bf16(acc[mi][ni][r]);
  }
}

// ---------------------------------------------------------------------------
// Kernel 2: scores, stored transposed: sT[b][j][i] = sum_d k[b,j,d]*q[b,i,d],
// then mask (mask[b,i,j]) and scale by 1/sqrt(2048) (mask BEFORE scale).
// Staging is a pure bf16 copy -> Tensor Data Mover, double-buffered: wave 0
// issues the next tile's tensor_load_to_lds while all waves run WMMAs.
// ---------------------------------------------------------------------------
__global__ __launch_bounds__(256)
void scores_kernel(const unsigned short* __restrict__ k_bf,
                   const unsigned short* __restrict__ q_bf,
                   const unsigned char*  __restrict__ mask,
                   float* __restrict__ sT) {
  const int i0 = blockIdx.x * 128;   // N dim = query index i
  const int j0 = blockIdx.y * 128;   // M dim = key index j
  const int b  = blockIdx.z;

  __shared__ __align__(16) unsigned short ldsA[2][128 * LDSK];  // k rows [j][d]
  __shared__ __align__(16) unsigned short ldsB[2][128 * LDSK];  // q rows [i][d]

  const int tid  = threadIdx.x;
  const int lane = tid & 31;
  const int wave = tid >> 5;
  const int wm   = wave & 1;
  const int wn   = wave >> 1;

  const unsigned short* gA = k_bf + (size_t)(b * SEQ + j0) * DIMA;
  const unsigned short* gB = q_bf + (size_t)(b * SEQ + i0) * DIMA;

  v8f zero = {};
  v8f acc[4][2];
#pragma unroll
  for (int mi = 0; mi < 4; ++mi)
#pragma unroll
    for (int ni = 0; ni < 2; ++ni) acc[mi][ni] = zero;

  // Prologue: DMA the first K-tile into buffer 0.
  if (wave == 0) {
    tdm_load_tile(gA, (unsigned int)(uintptr_t)&ldsA[0][0]);
    tdm_load_tile(gB, (unsigned int)(uintptr_t)&ldsB[0][0]);
  }

  const int NSTEP = DIMA / 32;   // 64
  for (int step = 0; step < NSTEP; ++step) {
    const int buf = step & 1;
    if (wave == 0) __builtin_amdgcn_s_wait_tensorcnt((short)0);
    __syncthreads();   // buf is filled; everyone done reading buf^1 from step-1
    if (wave == 0 && step + 1 < NSTEP) {
      const int kn = (step + 1) * 32;
      tdm_load_tile(gA + kn, (unsigned int)(uintptr_t)&ldsA[buf ^ 1][0]);
      tdm_load_tile(gB + kn, (unsigned int)(uintptr_t)&ldsB[buf ^ 1][0]);
    }

    v16bf afr[4], bfr[2];
#pragma unroll
    for (int mi = 0; mi < 4; ++mi)
      afr[mi] = load_frag_A(&ldsA[buf][0] + (wm * 64 + mi * 16) * LDSK, lane);
#pragma unroll
    for (int ni = 0; ni < 2; ++ni)
      bfr[ni] = load_frag_B(&ldsB[buf][0] + (wn * 32 + ni * 16) * LDSK, lane);
#pragma unroll
    for (int mi = 0; mi < 4; ++mi)
#pragma unroll
      for (int ni = 0; ni < 2; ++ni)
        acc[mi][ni] = __builtin_amdgcn_wmma_f32_16x16x32_bf16(
            false, afr[mi], false, bfr[ni], (short)0, acc[mi][ni], false, false);
  }

  const float scale = 0.022097086912079608f;       // 1/sqrt(2048)
  const float negv  = -1000.0f * scale;            // masked fill, then scaled
  const int ib = i0 + wn * 32 + (lane & 15);
  const int jb = j0 + wm * 64 + ((lane >> 4) << 3);
#pragma unroll
  for (int mi = 0; mi < 4; ++mi)
#pragma unroll
    for (int ni = 0; ni < 2; ++ni)
#pragma unroll
      for (int r = 0; r < 8; ++r) {
        const int j = jb + mi * 16 + r;
        const int i = ib + ni * 16;
        const unsigned char mk = mask[((size_t)b * SEQ + i) * SEQ + j];
        const float val = mk ? acc[mi][ni][r] * scale : negv;
        sT[((size_t)b * SEQ + j) * SEQ + i] = val;
      }
}

// ---------------------------------------------------------------------------
// Kernel 3: softmax stats per (b,j) row of sT (softmax over i == axis 1).
// ---------------------------------------------------------------------------
__global__ __launch_bounds__(256)
void softmax_stats_kernel(const float* __restrict__ sT,
                          float* __restrict__ rowmax,
                          float* __restrict__ rowinv) {
  const int row = blockIdx.x;                 // b*SEQ + j
  const float* p = sT + (size_t)row * SEQ;
  const int tid = threadIdx.x;
  __shared__ float red[256];

  float m = -3.4e38f;
#pragma unroll
  for (int k = 0; k < 8; ++k) m = fmaxf(m, p[tid + k * 256]);
  red[tid] = m; __syncthreads();
  for (int s = 128; s > 0; s >>= 1) {
    if (tid < s) red[tid] = fmaxf(red[tid], red[tid + s]);
    __syncthreads();
  }
  m = red[0]; __syncthreads();

  float sum = 0.0f;
#pragma unroll
  for (int k = 0; k < 8; ++k) sum += __expf(p[tid + k * 256] - m);
  red[tid] = sum; __syncthreads();
  for (int s = 128; s > 0; s >>= 1) {
    if (tid < s) red[tid] += red[tid + s];
    __syncthreads();
  }
  if (tid == 0) { rowmax[row] = m; rowinv[row] = 1.0f / red[0]; }
}

// ---------------------------------------------------------------------------
// Kernel 4: out[b,i,j] = exp(sT[b,j,i]-max[b,j])*inv[b,j] * v[b,j,i],
// via 64x64 LDS transpose so both reads and writes are coalesced.
// ---------------------------------------------------------------------------
__global__ __launch_bounds__(256)
void out_kernel(const float* __restrict__ sT,
                const float* __restrict__ v,
                const float* __restrict__ rowmax,
                const float* __restrict__ rowinv,
                float* __restrict__ out) {
  const int i0 = blockIdx.x * 64;
  const int j0 = blockIdx.y * 64;
  const int b  = blockIdx.z;
  __shared__ float tile[64 * 65];
  const int tid = threadIdx.x;

#pragma unroll
  for (int p = 0; p < 16; ++p) {
    const int e  = tid + p * 256;   // 4096 elements
    const int jr = e >> 6;
    const int ic = e & 63;
    const int row = b * SEQ + j0 + jr;
    const float s  = sT[(size_t)row * SEQ + i0 + ic];
    const float vv = v[(size_t)row * DIMA + i0 + ic];
    tile[jr * 65 + ic] = __expf(s - rowmax[row]) * rowinv[row] * vv;
  }
  __syncthreads();
#pragma unroll
  for (int p = 0; p < 16; ++p) {
    const int e  = tid + p * 256;
    const int ir = e >> 6;
    const int jc = e & 63;
    out[((size_t)(b * SEQ) + i0 + ir) * SEQ + j0 + jc] = tile[jc * 65 + ir];
  }
}

// ---------------------------------------------------------------------------
extern "C" void kernel_launch(void* const* d_in, const int* in_sizes, int n_in,
                              void* d_out, int out_size, void* d_ws, size_t ws_size,
                              hipStream_t stream) {
  const float* x  = (const float*)d_in[0];
  const float* wq = (const float*)d_in[1];
  const float* wk = (const float*)d_in[2];
  const float* wv = (const float*)d_in[3];
  const unsigned char* mask = (const unsigned char*)d_in[4];
  float* out = (float*)d_out;

  // Workspace layout (all 256B-aligned offsets):
  //   q bf16   : 16384 x 2048 x 2B =  64 MB
  //   k bf16   : 16384 x 2048 x 2B =  64 MB
  //   v f32    : 16384 x 2048 x 4B = 128 MB
  //   sT f32   : 8 x 2048 x 2048x4 = 128 MB
  //   rowmax/rowinv : 2 x 64 KB
  char* ws = (char*)d_ws;
  unsigned short* q_bf  = (unsigned short*)(ws);
  unsigned short* k_bf  = (unsigned short*)(ws + (size_t)67108864);
  float*          v_f32 = (float*)(ws + (size_t)134217728);
  float*          sT    = (float*)(ws + (size_t)268435456);
  float*          rmax  = (float*)(ws + (size_t)402653184);
  float*          rinv  = (float*)(ws + (size_t)402653184 + 65536);

  qkv_proj_kernel<<<dim3(DIMA / 128, (BATCH * SEQ) / 128, 3), 256, 0, stream>>>(
      x, wq, wk, wv, q_bf, k_bf, v_f32);
  scores_kernel<<<dim3(SEQ / 128, SEQ / 128, BATCH), 256, 0, stream>>>(
      k_bf, q_bf, mask, sT);
  softmax_stats_kernel<<<dim3(BATCH * SEQ), 256, 0, stream>>>(sT, rmax, rinv);
  out_kernel<<<dim3(SEQ / 64, SEQ / 64, BATCH), 256, 0, stream>>>(
      sT, v_f32, rmax, rinv, out);
}
